// QKVAttention_41936060678240
// MI455X (gfx1250) — compile-verified, matrix operand order checked by reading, not run on hardware
//
#include <hip/hip_runtime.h>
#include <hip/hip_bf16.h>

typedef __attribute__((ext_vector_type(16))) __bf16 v16bf;
typedef __attribute__((ext_vector_type(8)))  __bf16 bf8v;
typedef __attribute__((ext_vector_type(4)))  __bf16 bf4v;
typedef __attribute__((ext_vector_type(8)))  float  v8f;
typedef __attribute__((ext_vector_type(4)))  float  f4;

#define WMMA_BF16(A, B, C) \
  __builtin_amdgcn_wmma_f32_16x16x32_bf16(false, (A), false, (B), (short)0, (C), false, false)

// Problem constants (from reference setup_inputs)
constexpr int BATCH = 8, MTOT = 2048, NTOT = 2048, DK = 128;
// 1/sqrt(128) * log2(e): fold scale + natural->base2 exp into scores
constexpr float SCALE_LOG2E = 0.12751454f;
// 1e32 * SCALE_LOG2E: additive mask for absent keys, already in log2 domain
constexpr float BIAS_BIG = 1.2751454e31f;

// Flash-attention forward, transposed-S formulation.
//   Per wave: 16 query rows.  Per block (4 waves): 64 query rows.
//   KV tiles of 64 keys staged fp32->bf16 into LDS each iteration.
//   S^T = K * Q^T + bias  (A = K tile, B = Q^T fragment, C seeded with the
//                          per-key additive mask)  -> C: lane = query col
//   O^T += V^T * P^T (A = V^T from LDS, B = P^T straight from C registers)
__global__ __launch_bounds__(128) void fa_fwd_kernel(
    const float* __restrict__ Q, const float* __restrict__ K,
    const float* __restrict__ V, const int* __restrict__ qpres,
    const int* __restrict__ kpres, float* __restrict__ Out) {
  const int tid  = threadIdx.x;
  const int lane = tid & 31;
  const int wave = tid >> 5;
  const int l15  = lane & 15;          // query col within tile / A-matrix row
  const int hb   = (lane >> 4) << 3;   // 0 or 8 : K-dim half select
  const int b    = blockIdx.x >> 5;    // 32 M-blocks per batch
  const int m0   = (blockIdx.x & 31) * 64 + wave * 16;

  // LDS layout (36096 B total, reused for output transpose in epilogue):
  //   Ks  : 64 keys x 136 bf16 (padded rows -> conflict-free b128 frags)
  //   Vt  : 128 d   x  72 bf16 (V transposed, padded)
  //   bias: 64 floats, (kp-1)*BIAS_BIG per key
  __shared__ __align__(16) unsigned char smem[36096];
  __bf16* Ks    = (__bf16*)(smem);           // 17408 B
  __bf16* Vt    = (__bf16*)(smem + 17408);   // 18432 B
  float*  biasS = (float*)(smem + 35840);    //   256 B

  // ---- Q fragments (B operand of S^T GEMM); fold q_pres * scale * log2e ----
  const float* Qrow   = Q + ((size_t)b * MTOT + (m0 + l15)) * DK;
  const float  qscale = SCALE_LOG2E * (float)qpres[b * MTOT + m0 + l15];
  v16bf qf[4];
#pragma unroll
  for (int c = 0; c < 4; ++c) {
    const f4* p0 = (const f4*)(Qrow + c * 32 + hb);
    const f4* p1 = (const f4*)(Qrow + c * 32 + 16 + hb);
    f4 x0 = p0[0], x1 = p0[1], y0 = p1[0], y1 = p1[1];
#pragma unroll
    for (int j = 0; j < 4; ++j) {
      qf[c][j]      = (__bf16)(x0[j] * qscale);
      qf[c][4 + j]  = (__bf16)(x1[j] * qscale);
      qf[c][8 + j]  = (__bf16)(y0[j] * qscale);
      qf[c][12 + j] = (__bf16)(y1[j] * qscale);
    }
  }

  v8f Ot[8];                       // O^T accumulators: 8 d-tiles of 16x16
#pragma unroll
  for (int d = 0; d < 8; ++d)
#pragma unroll
    for (int i = 0; i < 8; ++i) Ot[d][i] = 0.0f;
  float mrun = -3.0e38f, lrun = 0.0f;  // online-softmax state (log2 domain)

  for (int n0 = 0; n0 < NTOT; n0 += 64) {
    __syncthreads();  // previous iteration's readers done before restage
    if (tid < 64)
      biasS[tid] = (float)(kpres[b * NTOT + n0 + tid] - 1) * BIAS_BIG;
#pragma unroll 4
    for (int p = 0; p < 16; ++p) {
      int    li   = p * 128 + tid;
      int    key  = li >> 5;
      int    dv   = (li & 31) * 4;
      size_t goff = ((size_t)b * NTOT + (n0 + key)) * DK + dv;
      f4     kvec = *(const f4*)(K + goff);   // coalesced 512B/wave
      f4     vvec = *(const f4*)(V + goff);
      float  kpf  = (float)kpres[b * NTOT + n0 + key];
      bf4v   kb;
#pragma unroll
      for (int j = 0; j < 4; ++j) kb[j] = (__bf16)(kvec[j] * kpf);
      *(bf4v*)(Ks + key * 136 + dv) = kb;     // b64 store, row-major bf16
#pragma unroll
      for (int j = 0; j < 4; ++j)             // transpose V into [d][key]
        Vt[(dv + j) * 72 + key] = (__bf16)vvec[j];
    }
    __syncthreads();

    // ---- S^T tiles: 4 subtiles of 16 keys; key = n0 + 16t + (i + 8h) ----
    v8f S[4];
#pragma unroll
    for (int t = 0; t < 4; ++t) {
      // Seed the accumulator with the additive key mask: S = K*Q^T + bias.
      // (bias is per C-row = per key; identical across lanes in each half.)
      const f4* bp = (const f4*)(biasS + t * 16 + hb);
      f4  b0 = bp[0], b1 = bp[1];
      v8f acc;
#pragma unroll
      for (int i = 0; i < 4; ++i) { acc[i] = b0[i]; acc[4 + i] = b1[i]; }
#pragma unroll
      for (int c = 0; c < 4; ++c) {           // d-chunks of 32
        const __bf16* kr = Ks + (t * 16 + l15) * 136 + c * 32 + hb;
        bf8v  a0 = *(const bf8v*)(kr);
        bf8v  a1 = *(const bf8v*)(kr + 16);
        v16bf a  = __builtin_shufflevector(a0, a1, 0, 1, 2, 3, 4, 5, 6, 7, 8,
                                           9, 10, 11, 12, 13, 14, 15);
        acc = WMMA_BF16(a, qf[c], acc);
      }
      S[t] = acc;
    }

    // ---- online softmax over 64 keys (per query column m = lane&15) ----
    float pm = -3.0e38f;
#pragma unroll
    for (int t = 0; t < 4; ++t)
#pragma unroll
      for (int i = 0; i < 8; ++i) pm = fmaxf(pm, S[t][i]);
    pm = fmaxf(pm, __shfl_xor(pm, 16, 32));   // combine the two key halves
    float mnew = fmaxf(mrun, pm);
    float corr = __builtin_amdgcn_exp2f(mrun - mnew);
    mrun = mnew;

    float ps = 0.0f;
    v16bf pb[2];  // P^T in B-fragment layout: falls straight out of C layout
#pragma unroll
    for (int t = 0; t < 4; ++t)
#pragma unroll
      for (int i = 0; i < 8; ++i) {
        float pv = __builtin_amdgcn_exp2f(S[t][i] - mnew);
        ps += pv;
        pb[t >> 1][(t & 1) * 8 + i] = (__bf16)pv;
      }
    ps += __shfl_xor(ps, 16, 32);
    lrun = lrun * corr + ps;
#pragma unroll
    for (int d = 0; d < 8; ++d) Ot[d] *= corr;

    // ---- O^T += V^T * P^T ----
#pragma unroll
    for (int kc = 0; kc < 2; ++kc)
#pragma unroll
      for (int dt = 0; dt < 8; ++dt) {
        const __bf16* vr = Vt + (dt * 16 + l15) * 72 + kc * 32 + hb;
        bf8v  a0 = *(const bf8v*)(vr);
        bf8v  a1 = *(const bf8v*)(vr + 16);
        v16bf a  = __builtin_shufflevector(a0, a1, 0, 1, 2, 3, 4, 5, 6, 7, 8,
                                           9, 10, 11, 12, 13, 14, 15);
        Ot[dt] = WMMA_BF16(a, pb[kc], Ot[dt]);
      }
  }

  // ---- epilogue: normalize, transpose O^T back via LDS, coalesced store ----
  __syncthreads();  // all waves done with Ks/Vt before LDS reuse
  float*      Os = (float*)smem + wave * (16 * 132);  // [16 m][132 d padded]
  const float rl = 1.0f / lrun;
#pragma unroll
  for (int dt = 0; dt < 8; ++dt)
#pragma unroll
    for (int i = 0; i < 8; ++i)
      Os[l15 * 132 + dt * 16 + hb + i] = Ot[dt][i] * rl;
  __syncthreads();
  float* outBase = Out + ((size_t)b * MTOT + m0) * DK;
#pragma unroll
  for (int p = 0; p < 16; ++p) {
    int li = p * 32 + lane;
    int mr = li >> 5;
    int dv = (li & 31) * 4;
    f4  v  = *(const f4*)(Os + mr * 132 + dv);
    *(f4*)(outBase + mr * DK + dv) = v;   // coalesced float4 stores
  }
}

extern "C" void kernel_launch(void* const* d_in, const int* in_sizes, int n_in,
                              void* d_out, int out_size, void* d_ws,
                              size_t ws_size, hipStream_t stream) {
  (void)in_sizes; (void)n_in; (void)out_size; (void)d_ws; (void)ws_size;
  const float* Q  = (const float*)d_in[0];
  const float* K  = (const float*)d_in[1];
  const float* V  = (const float*)d_in[2];
  const int*   qp = (const int*)d_in[3];
  const int*   kp = (const int*)d_in[4];
  float*       out = (float*)d_out;
  // grid: 8 batches * 32 M-blocks (64 query rows each); block: 4 wave32s
  fa_fwd_kernel<<<dim3(BATCH * (MTOT / 64)), dim3(128), 0, stream>>>(
      Q, K, V, qp, kp, out);
}
